// TransformerEncoderDecoder_63668595196498
// MI455X (gfx1250) — matrix operand order repaired and checked by reading, not compiled, not run on hardware
//
#include <hip/hip_runtime.h>
#include <hip/hip_bf16.h>

typedef __attribute__((ext_vector_type(16))) __bf16 v16bf;
typedef __attribute__((ext_vector_type(8)))  float  v8f;

union FragBF { v16bf v; uint4 q[2]; __bf16 e[16]; };

// ---- WMMA fragment loaders (wave32, 16x16x32 bf16) ----
// A-matrix 16x32: lane<16 -> row=lane, K in {kk..kk+7, kk+16..kk+23}
//                 lane>=16 -> row=lane-16, K in {kk+8..kk+15, kk+24..kk+31}
__device__ __forceinline__ v16bf load_afrag(const __bf16* base, int ld, int kk) {
  int lane  = threadIdx.x & 31;
  int r     = lane & 15;
  int khalf = lane >> 4;
  const __bf16* p = base + (size_t)r * ld + kk + khalf * 8;
  FragBF f;
  f.q[0] = *(const uint4*)(p);
  f.q[1] = *(const uint4*)(p + 16);
  return f.v;
}

// B-matrix 32x16: lane<16 -> col=lane, K in {kk..kk+7, kk+16..kk+23} ; lane>=16 -> col=lane-16, K in {kk+8..kk+15, kk+24..kk+31}
// source rows indexed by column n, K contiguous (i.e. B^T row-major)
__device__ __forceinline__ v16bf load_bfrag(const __bf16* base, int ld, int kk) {
  int lane  = threadIdx.x & 31;
  int n     = lane & 15;
  int khalf = lane >> 4;
  const __bf16* p = base + (size_t)n * ld + kk + khalf * 8;
  FragBF f;
  f.q[0] = *(const uint4*)(p);
  f.q[1] = *(const uint4*)(p + 16);
  return f.v;
}

// ---- Generic GEMM: Y[M,N] = X_bf16[M,K] * W_f32[K,N] (+bias)(relu)(+resid_f32) ----
// grid = (M/128, N/64), block = 256 (8 waves, each wave: 16 rows x 64 cols).
// Weight chunk [64k x 64n] is staged through LDS (fp32 -> bf16, transposed to
// Wt[n][k], row stride 72 elems = 144B: 16B-aligned ds_load_b128, conflict-free),
// double-buffered; 8 WMMAs per barrier, B-fragments preloaded into independent
// registers so ds_load_b128s can run ahead of the WMMA issue.
__global__ __launch_bounds__(256) void gemm_bf16_kernel(
    const __bf16* __restrict__ X, const float* __restrict__ W,
    const float* __restrict__ bias, const float* __restrict__ resid,
    float* __restrict__ outF, __bf16* __restrict__ outB,
    int M, int N, int K, int relu)
{
  __shared__ __bf16 Wt[2][64][72];
  int tid  = threadIdx.x;
  int w    = tid >> 5;
  int lane = tid & 31;
  int m0 = blockIdx.x * 128 + w * 16;
  int n0 = blockIdx.y * 64;
  v8f acc[4] = {};
  const __bf16* Xb = X + (size_t)m0 * K;

  // cooperative stage of a 64k x 64n chunk -> LDS buffer bufi (transpose + cvt)
  auto stage = [&](int kk, int bufi) {
    int krow = tid >> 2;          // 0..63
    int ncol = (tid & 3) * 16;    // 0,16,32,48
    const float* p = W + (size_t)(kk + krow) * N + n0 + ncol;
    float4 f0 = *(const float4*)(p);
    float4 f1 = *(const float4*)(p + 4);
    float4 f2 = *(const float4*)(p + 8);
    float4 f3 = *(const float4*)(p + 12);
    __bf16 e[16] = { (__bf16)f0.x, (__bf16)f0.y, (__bf16)f0.z, (__bf16)f0.w,
                     (__bf16)f1.x, (__bf16)f1.y, (__bf16)f1.z, (__bf16)f1.w,
                     (__bf16)f2.x, (__bf16)f2.y, (__bf16)f2.z, (__bf16)f2.w,
                     (__bf16)f3.x, (__bf16)f3.y, (__bf16)f3.z, (__bf16)f3.w };
#pragma unroll
    for (int j = 0; j < 16; ++j) Wt[bufi][ncol + j][krow] = e[j];
  };

  stage(0, 0);
  int buf = 0;
  for (int kk = 0; kk < K; kk += 64) {
    __syncthreads();
    if (kk + 64 < K) stage(kk + 64, buf ^ 1);

    v16bf a0 = load_afrag(Xb, K, kk);
    v16bf a1 = load_afrag(Xb, K, kk + 32);

    v16bf b00 = load_bfrag(&Wt[buf][ 0][0], 72, 0);
    v16bf b10 = load_bfrag(&Wt[buf][16][0], 72, 0);
    v16bf b20 = load_bfrag(&Wt[buf][32][0], 72, 0);
    v16bf b30 = load_bfrag(&Wt[buf][48][0], 72, 0);
    acc[0] = __builtin_amdgcn_wmma_f32_16x16x32_bf16(false, a0, false, b00, (short)0, acc[0], false, false);
    acc[1] = __builtin_amdgcn_wmma_f32_16x16x32_bf16(false, a0, false, b10, (short)0, acc[1], false, false);
    acc[2] = __builtin_amdgcn_wmma_f32_16x16x32_bf16(false, a0, false, b20, (short)0, acc[2], false, false);
    acc[3] = __builtin_amdgcn_wmma_f32_16x16x32_bf16(false, a0, false, b30, (short)0, acc[3], false, false);

    v16bf b01 = load_bfrag(&Wt[buf][ 0][0], 72, 32);
    v16bf b11 = load_bfrag(&Wt[buf][16][0], 72, 32);
    v16bf b21 = load_bfrag(&Wt[buf][32][0], 72, 32);
    v16bf b31 = load_bfrag(&Wt[buf][48][0], 72, 32);
    acc[0] = __builtin_amdgcn_wmma_f32_16x16x32_bf16(false, a1, false, b01, (short)0, acc[0], false, false);
    acc[1] = __builtin_amdgcn_wmma_f32_16x16x32_bf16(false, a1, false, b11, (short)0, acc[1], false, false);
    acc[2] = __builtin_amdgcn_wmma_f32_16x16x32_bf16(false, a1, false, b21, (short)0, acc[2], false, false);
    acc[3] = __builtin_amdgcn_wmma_f32_16x16x32_bf16(false, a1, false, b31, (short)0, acc[3], false, false);

    buf ^= 1;
  }

  int laneHi = lane >> 4;
  int nlo    = lane & 15;
#pragma unroll
  for (int t = 0; t < 4; ++t) {
    int n = n0 + 16 * t + nlo;
    float bv = bias ? bias[n] : 0.0f;
#pragma unroll
    for (int r = 0; r < 8; ++r) {
      int m = m0 + r + 8 * laneHi;
      float val = acc[t][r] + bv;
      if (relu) val = fmaxf(val, 0.0f);
      size_t idx = (size_t)m * N + n;
      if (resid) val += resid[idx];
      if (outF) outF[idx] = val;
      if (outB) outB[idx] = (__bf16)val;
    }
  }
}

// ---- Attention: one block per (batch, head). S=128, DH=64, D=512 ----
// block = 256 threads (8 waves); wave w handles query rows [16w, 16w+16)
__global__ __launch_bounds__(256) void attn_kernel(
    const __bf16* __restrict__ Q, const __bf16* __restrict__ Kb,
    const __bf16* __restrict__ Vb, __bf16* __restrict__ O, int causal)
{
  __shared__ __bf16 Vt[64][136];       // V transposed: Vt[d][k]
  __shared__ __bf16 Pw[8][16][136];    // per-wave softmax probs, bf16

  int b = blockIdx.x >> 3;
  int h = blockIdx.x & 7;
  int tid = threadIdx.x;

  // stage V^T into LDS
  for (int idx = tid; idx < 128 * 64; idx += 256) {
    int k = idx >> 6, d = idx & 63;
    Vt[d][k] = Vb[((size_t)(b * 128 + k)) * 512 + h * 64 + d];
  }
  __syncthreads();

  int w = tid >> 5, lane = tid & 31;
  int laneHi = lane >> 4, nlo = lane & 15;
  int q0 = w * 16;
  const __bf16* Qbase = Q + ((size_t)(b * 128 + q0)) * 512 + h * 64;

  // S = Q * K^T  (8 key tiles of 16, K-dim = 64)
  v8f s[8] = {};
  for (int kk = 0; kk < 64; kk += 32) {
    v16bf a = load_afrag(Qbase, 512, kk);
#pragma unroll
    for (int t = 0; t < 8; ++t) {
      const __bf16* Kbase = Kb + ((size_t)(b * 128 + t * 16)) * 512 + h * 64;
      v16bf bb = load_bfrag(Kbase, 512, kk);
      s[t] = __builtin_amdgcn_wmma_f32_16x16x32_bf16(false, a, false, bb,
                                                     (short)0, s[t], false, false);
    }
  }
  // scale + causal mask
#pragma unroll
  for (int t = 0; t < 8; ++t) {
#pragma unroll
    for (int r = 0; r < 8; ++r) {
      float val = s[t][r] * 0.125f;   // 1/sqrt(64)
      if (causal) {
        int qi = q0 + r + 8 * laneHi;
        int ki = t * 16 + nlo;
        if (ki > qi) val = -1e30f;
      }
      s[t][r] = val;
    }
  }
  // softmax per row: a row's 16 columns live across one 16-lane half, x 8 tiles
#pragma unroll
  for (int r = 0; r < 8; ++r) {
    float mx = -1e30f;
#pragma unroll
    for (int t = 0; t < 8; ++t) mx = fmaxf(mx, s[t][r]);
    for (int m = 1; m < 16; m <<= 1) mx = fmaxf(mx, __shfl_xor(mx, m, 32));
    float sum = 0.0f;
#pragma unroll
    for (int t = 0; t < 8; ++t) { float e = __expf(s[t][r] - mx); s[t][r] = e; sum += e; }
    for (int m = 1; m < 16; m <<= 1) sum += __shfl_xor(sum, m, 32);
    float inv = 1.0f / sum;
#pragma unroll
    for (int t = 0; t < 8; ++t) s[t][r] *= inv;
  }
  // C-layout -> A-layout via per-wave LDS (bf16); LDS ops in-order within wave
#pragma unroll
  for (int t = 0; t < 8; ++t)
#pragma unroll
    for (int r = 0; r < 8; ++r)
      Pw[w][r + 8 * laneHi][t * 16 + nlo] = (__bf16)s[t][r];

  // O = P * V  (4 d-tiles of 16, K-dim = 128 keys)
  v8f o[4] = {};
  for (int kk = 0; kk < 128; kk += 32) {
    v16bf a = load_afrag(&Pw[w][0][0], 136, kk);
    v16bf bb0 = load_bfrag(&Vt[ 0][0], 136, kk);
    v16bf bb1 = load_bfrag(&Vt[16][0], 136, kk);
    v16bf bb2 = load_bfrag(&Vt[32][0], 136, kk);
    v16bf bb3 = load_bfrag(&Vt[48][0], 136, kk);
    o[0] = __builtin_amdgcn_wmma_f32_16x16x32_bf16(false, a, false, bb0, (short)0, o[0], false, false);
    o[1] = __builtin_amdgcn_wmma_f32_16x16x32_bf16(false, a, false, bb1, (short)0, o[1], false, false);
    o[2] = __builtin_amdgcn_wmma_f32_16x16x32_bf16(false, a, false, bb2, (short)0, o[2], false, false);
    o[3] = __builtin_amdgcn_wmma_f32_16x16x32_bf16(false, a, false, bb3, (short)0, o[3], false, false);
  }
#pragma unroll
  for (int dt = 0; dt < 4; ++dt)
#pragma unroll
    for (int r = 0; r < 8; ++r) {
      int m = q0 + r + 8 * laneHi;
      int d = dt * 16 + nlo;
      O[((size_t)(b * 128 + m)) * 512 + h * 64 + d] = (__bf16)o[dt][r];
    }
}

// ---- LayerNorm over D=512, one row per block (256 threads, 2 elems each) ----
__global__ __launch_bounds__(256) void ln_kernel(
    const float* __restrict__ x, const float* __restrict__ g,
    const float* __restrict__ bta, float* __restrict__ outF,
    __bf16* __restrict__ outB)
{
  __shared__ float p1[8], p2[8];
  int row = blockIdx.x;
  const float* xr = x + (size_t)row * 512;
  int c0 = threadIdx.x, c1 = threadIdx.x + 256;
  float a0 = xr[c0], a1 = xr[c1];
  int w = threadIdx.x >> 5, lane = threadIdx.x & 31;

  float s = a0 + a1;
  for (int m = 1; m < 32; m <<= 1) s += __shfl_xor(s, m, 32);
  if (lane == 0) p1[w] = s;
  __syncthreads();
  float tot = 0.0f;
#pragma unroll
  for (int i = 0; i < 8; ++i) tot += p1[i];
  float mean = tot * (1.0f / 512.0f);

  float d0 = a0 - mean, d1 = a1 - mean;
  float vs = d0 * d0 + d1 * d1;
  for (int m = 1; m < 32; m <<= 1) vs += __shfl_xor(vs, m, 32);
  if (lane == 0) p2[w] = vs;
  __syncthreads();
  float vtot = 0.0f;
#pragma unroll
  for (int i = 0; i < 8; ++i) vtot += p2[i];
  float inv = rsqrtf(vtot * (1.0f / 512.0f) + 1e-6f);

  float y0 = g[c0] * d0 * inv + bta[c0];
  float y1 = g[c1] * d1 * inv + bta[c1];
  size_t base = (size_t)row * 512;
  if (outF) { outF[base + c0] = y0; outF[base + c1] = y1; }
  if (outB) { outB[base + c0] = (__bf16)y0; outB[base + c1] = (__bf16)y1; }
}

// ---- embedding * sqrt(D) + sinusoidal PE; one block per (b,s) ----
__global__ __launch_bounds__(256) void embed_kernel(
    const int* __restrict__ tok, const float* __restrict__ emb,
    float* __restrict__ out)
{
  int rs = blockIdx.x;          // b*S + s
  int spos = rs & 127;
  int t = tok[rs];
  for (int d = threadIdx.x; d < 512; d += 256) {
    int i2 = (d >> 1) * 2;
    float div = __expf((float)i2 * (-9.210340371976184f / 512.0f)); // -ln(10000)/D
    float ang = (float)spos * div;
    float pe = (d & 1) ? __cosf(ang) : __sinf(ang);
    out[(size_t)rs * 512 + d] = emb[(size_t)t * 512 + d] * 22.62741699796952f + pe;
  }
}

extern "C" void kernel_launch(void* const* d_in, const int* in_sizes, int n_in,
                              void* d_out, int out_size, void* d_ws, size_t ws_size,
                              hipStream_t stream) {
  const int*   src      = (const int*)d_in[0];
  const int*   tgt      = (const int*)d_in[1];
  const float* src_emb  = (const float*)d_in[2];
  const float* tgt_emb  = (const float*)d_in[3];
  const float* enc_qkv_w = (const float*)d_in[4];
  const float* enc_qkv_b = (const float*)d_in[5];
  const float* enc_o_w   = (const float*)d_in[6];
  const float* enc_o_b   = (const float*)d_in[7];
  const float* enc_ff_w1 = (const float*)d_in[8];
  const float* enc_ff_b1 = (const float*)d_in[9];
  const float* enc_ff_w2 = (const float*)d_in[10];
  const float* enc_ff_b2 = (const float*)d_in[11];
  const float* enc_ln_g  = (const float*)d_in[12];
  const float* enc_ln_b  = (const float*)d_in[13];
  const float* enc_fg    = (const float*)d_in[14];
  const float* enc_fb    = (const float*)d_in[15];
  const float* dec_sqkv_w = (const float*)d_in[16];
  const float* dec_sqkv_b = (const float*)d_in[17];
  const float* dec_so_w   = (const float*)d_in[18];
  const float* dec_so_b   = (const float*)d_in[19];
  const float* dec_cqkv_w = (const float*)d_in[20];
  const float* dec_cqkv_b = (const float*)d_in[21];
  const float* dec_co_w   = (const float*)d_in[22];
  const float* dec_co_b   = (const float*)d_in[23];
  const float* dec_ff_w1  = (const float*)d_in[24];
  const float* dec_ff_b1  = (const float*)d_in[25];
  const float* dec_ff_w2  = (const float*)d_in[26];
  const float* dec_ff_b2  = (const float*)d_in[27];
  const float* dec_ln_g   = (const float*)d_in[28];
  const float* dec_ln_b   = (const float*)d_in[29];
  const float* dec_fg     = (const float*)d_in[30];
  const float* dec_fb     = (const float*)d_in[31];

  const int R = 1024;  // B*S = B*T rows
  char* ws = (char*)d_ws;
  float*  x    = (float*)(ws + 0);                     // 2 MB encoder stream
  float*  y    = (float*)(ws + (1u << 21));            // 2 MB decoder stream
  __bf16* h    = (__bf16*)(ws + (2u << 21));           // 1 MB LN output
  __bf16* q    = (__bf16*)(ws + (2u << 21) + 1 * 1048576);
  __bf16* k    = (__bf16*)(ws + (2u << 21) + 2 * 1048576);
  __bf16* v    = (__bf16*)(ws + (2u << 21) + 3 * 1048576);
  __bf16* attn = (__bf16*)(ws + (2u << 21) + 4 * 1048576);
  __bf16* ff1  = (__bf16*)(ws + (2u << 21) + 5 * 1048576); // 4 MB
  __bf16* memb = (__bf16*)(ws + (2u << 21) + 9 * 1048576);

  dim3 blk(256);
  dim3 gemmDD(8, 8);    // M=1024, N=512
  dim3 gemmDF(8, 32);   // M=1024, N=2048

  auto gemm = [&](const __bf16* X, const float* W, const float* B,
                  const float* res, float* oF, __bf16* oB,
                  int M, int N, int K, int relu, dim3 grid) {
    gemm_bf16_kernel<<<grid, blk, 0, stream>>>(X, W, B, res, oF, oB, M, N, K, relu);
  };

  // ======== Encoder ========
  embed_kernel<<<R, blk, 0, stream>>>(src, src_emb, x);
  for (int i = 0; i < 6; ++i) {
    const float* Wqkv = enc_qkv_w + (size_t)(i * 3) * 512 * 512;
    const float* Bqkv = enc_qkv_b + (size_t)(i * 3) * 512;
    ln_kernel<<<R, blk, 0, stream>>>(x, enc_ln_g + (size_t)(i * 2) * 512,
                                     enc_ln_b + (size_t)(i * 2) * 512, nullptr, h);
    gemm(h, Wqkv,                 Bqkv,         nullptr, nullptr, q, R, 512, 512, 0, gemmDD);
    gemm(h, Wqkv + 512 * 512,     Bqkv + 512,   nullptr, nullptr, k, R, 512, 512, 0, gemmDD);
    gemm(h, Wqkv + 2 * 512 * 512, Bqkv + 1024,  nullptr, nullptr, v, R, 512, 512, 0, gemmDD);
    attn_kernel<<<64, blk, 0, stream>>>(q, k, v, attn, 0);
    gemm(attn, enc_o_w + (size_t)i * 512 * 512, enc_o_b + (size_t)i * 512,
         x, x, nullptr, R, 512, 512, 0, gemmDD);
    ln_kernel<<<R, blk, 0, stream>>>(x, enc_ln_g + (size_t)(i * 2 + 1) * 512,
                                     enc_ln_b + (size_t)(i * 2 + 1) * 512, nullptr, h);
    gemm(h, enc_ff_w1 + (size_t)i * 512 * 2048, enc_ff_b1 + (size_t)i * 2048,
         nullptr, nullptr, ff1, R, 2048, 512, 1, gemmDF);
    gemm(ff1, enc_ff_w2 + (size_t)i * 2048 * 512, enc_ff_b2 + (size_t)i * 512,
         x, x, nullptr, R, 512, 2048, 0, gemmDD);
  }
  ln_kernel<<<R, blk, 0, stream>>>(x, enc_fg, enc_fb, nullptr, memb);

  // ======== Decoder ========
  embed_kernel<<<R, blk, 0, stream>>>(tgt, tgt_emb, y);
  for (int i = 0; i < 6; ++i) {
    const float* Ws = dec_sqkv_w + (size_t)(i * 3) * 512 * 512;
    const float* Bs = dec_sqkv_b + (size_t)(i * 3) * 512;
    const float* Wc = dec_cqkv_w + (size_t)(i * 3) * 512 * 512;
    const float* Bc = dec_cqkv_b + (size_t)(i * 3) * 512;
    // self-attention (causal)
    ln_kernel<<<R, blk, 0, stream>>>(y, dec_ln_g + (size_t)(i * 3) * 512,
                                     dec_ln_b + (size_t)(i * 3) * 512, nullptr, h);
    gemm(h, Ws,                 Bs,         nullptr, nullptr, q, R, 512, 512, 0, gemmDD);
    gemm(h, Ws + 512 * 512,     Bs + 512,   nullptr, nullptr, k, R, 512, 512, 0, gemmDD);
    gemm(h, Ws + 2 * 512 * 512, Bs + 1024,  nullptr, nullptr, v, R, 512, 512, 0, gemmDD);
    attn_kernel<<<64, blk, 0, stream>>>(q, k, v, attn, 1);
    gemm(attn, dec_so_w + (size_t)i * 512 * 512, dec_so_b + (size_t)i * 512,
         y, y, nullptr, R, 512, 512, 0, gemmDD);
    // cross-attention (K/V from encoder memory)
    ln_kernel<<<R, blk, 0, stream>>>(y, dec_ln_g + (size_t)(i * 3 + 1) * 512,
                                     dec_ln_b + (size_t)(i * 3 + 1) * 512, nullptr, h);
    gemm(h,    Wc,                 Bc,         nullptr, nullptr, q, R, 512, 512, 0, gemmDD);
    gemm(memb, Wc + 512 * 512,     Bc + 512,   nullptr, nullptr, k, R, 512, 512, 0, gemmDD);
    gemm(memb, Wc + 2 * 512 * 512, Bc + 1024,  nullptr, nullptr, v, R, 512, 512, 0, gemmDD);
    attn_kernel<<<64, blk, 0, stream>>>(q, k, v, attn, 0);
    gemm(attn, dec_co_w + (size_t)i * 512 * 512, dec_co_b + (size_t)i * 512,
         y, y, nullptr, R, 512, 512, 0, gemmDD);
    // feed-forward
    ln_kernel<<<R, blk, 0, stream>>>(y, dec_ln_g + (size_t)(i * 3 + 2) * 512,
                                     dec_ln_b + (size_t)(i * 3 + 2) * 512, nullptr, h);
    gemm(h, dec_ff_w1 + (size_t)i * 512 * 2048, dec_ff_b1 + (size_t)i * 2048,
         nullptr, nullptr, ff1, R, 2048, 512, 1, gemmDF);
    gemm(ff1, dec_ff_w2 + (size_t)i * 2048 * 512, dec_ff_b2 + (size_t)i * 512,
         y, y, nullptr, R, 512, 2048, 0, gemmDD);
  }
  ln_kernel<<<R, blk, 0, stream>>>(y, dec_fg, dec_fb, (float*)d_out, nullptr);
}
